// HexPlane_44272522887574
// MI455X (gfx1250) — matrix operand order repaired and checked by reading, not compiled, not run on hardware
//
#include <hip/hip_runtime.h>

typedef _Float16 v16h __attribute__((ext_vector_type(16)));
typedef _Float16 v8h  __attribute__((ext_vector_type(8)));
typedef float    v8f  __attribute__((ext_vector_type(8)));
typedef float    v4f  __attribute__((ext_vector_type(4)));

#define GG   128      // label plane H=W
#define TT   64       // deform plane H=W
#define CC   24       // label channels per plane
#define CD   32       // deform channels
#define PP   16       // number of parts
#define KPAD 96       // 72 label feats padded to 3x32
#define NF   144      // 9*PP deform outputs
#define TAU_INV 10.0f

// workspace (float) layout: channel-last transposed planes
#define TL_SIZE (3*GG*GG*CC)   // 1,179,648 floats
#define TD_SIZE (TT*TT*CD)     //   131,072 floats

__global__ __launch_bounds__(256) void hex_prep(
    const float* __restrict__ p0, const float* __restrict__ p1,
    const float* __restrict__ p2, const float* __restrict__ dp,
    float* __restrict__ ws)
{
    int idx = blockIdx.x * blockDim.x + threadIdx.x;
    if (idx < TL_SIZE) {
        int c   = idx % CC;
        int pix = (idx / CC) % (GG*GG);
        int pl  = idx / (CC*GG*GG);
        const float* src = (pl == 0) ? p0 : ((pl == 1) ? p1 : p2);
        ws[idx] = src[c*GG*GG + pix];
    } else if (idx < TL_SIZE + TD_SIZE) {
        int j   = idx - TL_SIZE;
        int c   = j % CD;
        int pix = j / CD;
        ws[TL_SIZE + j] = dp[c*TT*TT + pix];
    }
}

struct __align__(16) HexSmem {
    union {
        _Float16 Bd[9*32*16];      // deform B operands [tile][lane][16]  (9216 B)
        float    logits[64*PP];    // label logits, used after Bd is dead (4096 B)
    } u;
    _Float16 Bl[3*32*16];          // label B operands [chunk][lane][16]  (3072 B)
    _Float16 Astg[64*KPAD];        // per-point feature staging (f16)    (12288 B)
    float    dout[64*NF];          // deform GEMM output (f32)           (36864 B)
};                                 // total 61,440 B

// A operand, contiguous K layout (matches B):
// lane l holds M=l%16; halves h -> K=h+base, base=16 if l>=16
__device__ inline v16h make_a(const _Float16* stg, int row, int kbase, int lane)
{
    int b = (lane & 16) ? 16 : 0;
    const _Float16* p = stg + row*KPAD + kbase + b;
    v8h lo = *(const v8h*)(p);
    v8h hi = *(const v8h*)(p + 8);
    v16h a;
#pragma unroll
    for (int i = 0; i < 8; ++i) { a[i] = lo[i]; a[i+8] = hi[i]; }
    return a;
}

__device__ inline v16h make_b(const _Float16* bops, int lane)
{
    const _Float16* p = bops + lane*16;
    v8h lo = *(const v8h*)(p);
    v8h hi = *(const v8h*)(p + 8);
    v16h b;
#pragma unroll
    for (int i = 0; i < 8; ++i) { b[i] = lo[i]; b[i+8] = hi[i]; }
    return b;
}

__global__ __launch_bounds__(64) void hex_main(
    const float* __restrict__ rays,  const float* __restrict__ scales,
    const float* __restrict__ rots,  const float* __restrict__ opac,
    const float* __restrict__ shs,   const float* __restrict__ timef,
    const float* __restrict__ labelW, const float* __restrict__ deformW,
    const float* __restrict__ ws, float* __restrict__ out, int n)
{
    __shared__ HexSmem sm;
    const int tid  = threadIdx.x;
    const int lane = tid & 31;
    const int wv   = tid >> 5;
    const int pt   = blockIdx.x * 64 + tid;
    const int ptc  = min(pt, n - 1);

    const float* tl = ws;             // [3][G*G][24]
    const float* td = ws + TL_SIZE;   // [T*T][32]

    // ---- build f16 B operands in LDS (B layout: lanes 0-15 K=0-15, 16-31 K=16-31)
    for (int idx = tid; idx < 9*512; idx += 64) {
        int t = idx >> 9, r = idx & 511, ln = r >> 4, h = r & 15;
        int K  = h + ((ln & 16) ? 16 : 0);
        int Nn = ln & 15;
        sm.u.Bd[idx] = (_Float16)deformW[(t*16 + Nn)*CD + K];
    }
    for (int idx = tid; idx < 3*512; idx += 64) {
        int ch = idx >> 9, r = idx & 511, ln = r >> 4, h = r & 15;
        int K  = ch*32 + h + ((ln & 16) ? 16 : 0);
        int Nn = ln & 15;
        sm.Bl[idx] = (K < 72) ? (_Float16)labelW[Nn*72 + K] : (_Float16)0.0f;
    }

    // ---- phase 1: deform-plane bilinear (x fixed at 31.5) -> Astg[tid][0..31]
    {
        float t  = timef[ptc];
        float gy = t * (float)(TT - 1);   // ((2t-1)+1)*0.5*(T-1)
        float y0f = floorf(gy);
        float wy  = gy - y0f;
        int   y0  = (int)y0f;
        float v0 = (y0 >= 0 && y0 <= TT-1) ? 1.0f : 0.0f;
        float v1 = (y0+1 >= 0 && y0+1 <= TT-1) ? 1.0f : 0.0f;
        float w0 = 0.5f * (1.0f - wy) * v0;
        float w1 = 0.5f * wy * v1;
        int y0c = min(max(y0, 0), TT-1);
        int y1c = min(max(y0 + 1, 0), TT-1);
        const float* r0 = td + (y0c*TT + 31)*CD;   // taps (y0,31),(y0,32) contiguous
        const float* r1 = td + (y1c*TT + 31)*CD;
        _Float16* dst = sm.Astg + tid*KPAD;
#pragma unroll
        for (int c = 0; c < CD; c += 4) {
            v4f a0 = *(const v4f*)(r0 + c);
            v4f a1 = *(const v4f*)(r0 + CD + c);
            v4f b0 = *(const v4f*)(r1 + c);
            v4f b1 = *(const v4f*)(r1 + CD + c);
#pragma unroll
            for (int i = 0; i < 4; ++i)
                dst[c+i] = (_Float16)(w0*(a0[i]+a1[i]) + w1*(b0[i]+b1[i]));
        }
    }
    __syncthreads();

    // ---- phase 2: deform GEMM  D[32pts x 144] = A[32x32] * Bd[32x144]
    // Prefetch all 9 B tiles into registers so no LDS-source waits sit between WMMAs;
    // D-stores only read acc regs at issue, so the WMMA stream is not dscnt-serialized.
    {
        v16h bd[9];
#pragma unroll
        for (int t = 0; t < 9; ++t) bd[t] = make_b(sm.u.Bd + t*512, lane);
        const int m0 = (lane & 16) ? 8 : 0;
        const int nn = lane & 15;
#pragma unroll
        for (int hf = 0; hf < 2; ++hf) {
            int mb = wv*32 + hf*16;
            v16h a = make_a(sm.Astg, mb + (lane & 15), 0, lane);
#pragma unroll
            for (int t = 0; t < 9; ++t) {
                v8f acc = {};
                acc = __builtin_amdgcn_wmma_f32_16x16x32_f16(
                        false, a, false, bd[t], (short)0, acc, false, false);
#pragma unroll
                for (int r = 0; r < 8; ++r)
                    sm.dout[(mb + m0 + r)*NF + t*16 + nn] = acc[r];
            }
        }
    }
    __syncthreads();

    // ---- phase 3: label-plane bilinear (3 planes x 24ch) -> Astg[tid][0..71], pad to 96
    float px = rays[ptc*3 + 0], py = rays[ptc*3 + 1], pz = rays[ptc*3 + 2];
    {
        float xyz[3] = { -px, -py, -pz };      // normalize_aabb reduces to negation
        const int ax0[3] = {0, 0, 1};
        const int ax1[3] = {1, 2, 2};
        _Float16* dst = sm.Astg + tid*KPAD;
#pragma unroll
        for (int pl = 0; pl < 3; ++pl) {
            float gx = (xyz[ax0[pl]] + 1.0f) * 0.5f * (float)(GG-1);
            float gy = (xyz[ax1[pl]] + 1.0f) * 0.5f * (float)(GG-1);
            float x0f = floorf(gx), y0f = floorf(gy);
            float wx = gx - x0f, wyv = gy - y0f;
            int x0 = (int)x0f, y0 = (int)y0f;
            float f[CC];
#pragma unroll
            for (int c = 0; c < CC; ++c) f[c] = 0.0f;
#pragma unroll
            for (int tap = 0; tap < 4; ++tap) {
                int xs = x0 + (tap & 1), ys = y0 + (tap >> 1);
                float w = ((tap & 1) ? wx : 1.0f - wx) *
                          ((tap >> 1) ? wyv : 1.0f - wyv);
                w *= (xs >= 0 && xs <= GG-1 && ys >= 0 && ys <= GG-1) ? 1.0f : 0.0f;
                int xc = min(max(xs, 0), GG-1), yc = min(max(ys, 0), GG-1);
                const float* src = tl + ((size_t)pl*GG*GG + (size_t)yc*GG + xc)*CC;
#pragma unroll
                for (int c = 0; c < CC; c += 4) {
                    v4f v = *(const v4f*)(src + c);
#pragma unroll
                    for (int i = 0; i < 4; ++i) f[c+i] += w * v[i];
                }
            }
#pragma unroll
            for (int c = 0; c < CC; ++c) dst[pl*CC + c] = (_Float16)f[c];
        }
#pragma unroll
        for (int k = 72; k < KPAD; ++k) dst[k] = (_Float16)0.0f;
    }
    __syncthreads();

    // ---- phase 4: label GEMM  logits[32pts x 16] = A[32x96] * Bl[96x16]
    {
        v16h bl[3];
#pragma unroll
        for (int kc = 0; kc < 3; ++kc) bl[kc] = make_b(sm.Bl + kc*512, lane);
        const int m0 = (lane & 16) ? 8 : 0;
        const int nn = lane & 15;
#pragma unroll
        for (int hf = 0; hf < 2; ++hf) {
            int mb = wv*32 + hf*16;
            int rowA = mb + (lane & 15);
            v16h a0 = make_a(sm.Astg, rowA, 0,  lane);
            v16h a1 = make_a(sm.Astg, rowA, 32, lane);
            v16h a2 = make_a(sm.Astg, rowA, 64, lane);
            v8f acc = {};
            acc = __builtin_amdgcn_wmma_f32_16x16x32_f16(
                    false, a0, false, bl[0], (short)0, acc, false, false);
            acc = __builtin_amdgcn_wmma_f32_16x16x32_f16(
                    false, a1, false, bl[1], (short)0, acc, false, false);
            acc = __builtin_amdgcn_wmma_f32_16x16x32_f16(
                    false, a2, false, bl[2], (short)0, acc, false, false);
#pragma unroll
            for (int r = 0; r < 8; ++r)
                sm.u.logits[(mb + m0 + r)*PP + nn] = acc[r];   // Bd dead, alias ok
        }
    }
    __syncthreads();

    // ---- phase 5: softmax, 6d->matrix blend, quaternion, outputs (per-lane, own point)
    {
        float lg[PP];
#pragma unroll
        for (int p = 0; p < PP; ++p) lg[p] = sm.u.logits[tid*PP + p];
        float mx = lg[0];
#pragma unroll
        for (int p = 1; p < PP; ++p) mx = fmaxf(mx, lg[p]);
        float wgt[PP], wsum = 0.0f;
#pragma unroll
        for (int p = 0; p < PP; ++p) { wgt[p] = __expf((lg[p] - mx) * TAU_INV); wsum += wgt[p]; }
        float winv = 1.0f / wsum;
#pragma unroll
        for (int p = 0; p < PP; ++p) wgt[p] *= winv;

        float R[9] = {0,0,0,0,0,0,0,0,0}, tr[3] = {0,0,0};
        for (int p = 0; p < PP; ++p) {
            const float* d = sm.dout + tid*NF + p*9;
            float a1x=d[0], a1y=d[1], a1z=d[2], a2x=d[3], a2y=d[4], a2z=d[5];
            float n1 = 1.0f / sqrtf(a1x*a1x + a1y*a1y + a1z*a1z);
            float b1x=a1x*n1, b1y=a1y*n1, b1z=a1z*n1;
            float dp2 = b1x*a2x + b1y*a2y + b1z*a2z;
            float c2x=a2x-dp2*b1x, c2y=a2y-dp2*b1y, c2z=a2z-dp2*b1z;
            float n2 = 1.0f / sqrtf(c2x*c2x + c2y*c2y + c2z*c2z);
            float b2x=c2x*n2, b2y=c2y*n2, b2z=c2z*n2;
            float b3x = b1y*b2z - b1z*b2y;
            float b3y = b1z*b2x - b1x*b2z;
            float b3z = b1x*b2y - b1y*b2x;
            float w = wgt[p];
            R[0]+=w*b1x; R[1]+=w*b1y; R[2]+=w*b1z;
            R[3]+=w*b2x; R[4]+=w*b2y; R[5]+=w*b2z;
            R[6]+=w*b3x; R[7]+=w*b3y; R[8]+=w*b3z;
            tr[0]+=w*d[6]; tr[1]+=w*d[7]; tr[2]+=w*d[8];
        }

        float m00=R[0], m01=R[1], m02=R[2], m10=R[3], m11=R[4], m12=R[5],
              m20=R[6], m21=R[7], m22=R[8];
        float qa0 = sqrtf(fmaxf(1.0f + m00 + m11 + m22, 0.0f));
        float qa1 = sqrtf(fmaxf(1.0f + m00 - m11 - m22, 0.0f));
        float qa2 = sqrtf(fmaxf(1.0f - m00 + m11 - m22, 0.0f));
        float qa3 = sqrtf(fmaxf(1.0f - m00 - m11 + m22, 0.0f));
        int   idx = 0; float best = qa0;
        if (qa1 > best) { best = qa1; idx = 1; }
        if (qa2 > best) { best = qa2; idx = 2; }
        if (qa3 > best) { best = qa3; idx = 3; }
        float qw, qx, qy, qz, qa;
        if (idx == 0)      { qa=qa0; qw=qa0*qa0; qx=m21-m12;  qy=m02-m20;  qz=m10-m01; }
        else if (idx == 1) { qa=qa1; qw=m21-m12; qx=qa1*qa1;  qy=m10+m01;  qz=m02+m20; }
        else if (idx == 2) { qa=qa2; qw=m02-m20; qx=m10+m01;  qy=qa2*qa2;  qz=m12+m21; }
        else               { qa=qa3; qw=m10-m01; qx=m20+m02;  qy=m21+m12;  qz=qa3*qa3; }
        float sc = 1.0f / (2.0f * fmaxf(qa, 0.1f));
        qw*=sc; qx*=sc; qy*=sc; qz*=sc;

        float ox = m00*px + m01*py + m02*pz + tr[0];
        float oy = m10*px + m11*py + m12*pz + tr[1];
        float oz = m20*px + m21*py + m22*pz + tr[2];

        float bw = rots[ptc*4+0], bx = rots[ptc*4+1], by = rots[ptc*4+2], bz = rots[ptc*4+3];
        float rw = qw*bw - qx*bx - qy*by - qz*bz;
        float rx = qw*bx + qx*bw + qy*bz - qz*by;
        float ry = qw*by - qx*bz + qy*bw + qz*bx;
        float rz = qw*bz + qx*by - qy*bx + qz*bw;
        if (rw < 0.0f) { rw=-rw; rx=-rx; ry=-ry; rz=-rz; }

        if (pt < n) {
            float* pts_o = out;
            float* sc_o  = out + (size_t)3*n;
            float* rot_o = out + (size_t)6*n;
            float* op_o  = out + (size_t)10*n;
            float* shs_o = out + (size_t)11*n;
            pts_o[pt*3+0]=ox; pts_o[pt*3+1]=oy; pts_o[pt*3+2]=oz;
            sc_o[pt*3+0]=scales[pt*3+0]; sc_o[pt*3+1]=scales[pt*3+1]; sc_o[pt*3+2]=scales[pt*3+2];
            rot_o[pt*4+0]=rw; rot_o[pt*4+1]=rx; rot_o[pt*4+2]=ry; rot_o[pt*4+3]=rz;
            op_o[pt] = opac[pt];
#pragma unroll
            for (int i = 0; i < 12; ++i) {
                v4f v = *(const v4f*)(shs + (size_t)pt*48 + i*4);
                *(v4f*)(shs_o + (size_t)pt*48 + i*4) = v;
            }
        }
    }
}

extern "C" void kernel_launch(void* const* d_in, const int* in_sizes, int n_in,
                              void* d_out, int out_size, void* d_ws, size_t ws_size,
                              hipStream_t stream)
{
    const float* rays    = (const float*)d_in[0];
    const float* scales  = (const float*)d_in[1];
    const float* rotsv   = (const float*)d_in[2];
    const float* opac    = (const float*)d_in[3];
    const float* shs     = (const float*)d_in[4];
    const float* timef   = (const float*)d_in[5];
    const float* p0      = (const float*)d_in[6];
    const float* p1      = (const float*)d_in[7];
    const float* p2      = (const float*)d_in[8];
    const float* dp      = (const float*)d_in[9];
    const float* labelW  = (const float*)d_in[10];
    const float* deformW = (const float*)d_in[11];
    float* ws  = (float*)d_ws;
    float* out = (float*)d_out;
    const int n = in_sizes[5];   // time_feature flat count == N

    const int prep_total = TL_SIZE + TD_SIZE;
    hex_prep<<<(prep_total + 255)/256, 256, 0, stream>>>(p0, p1, p2, dp, ws);
    hex_main<<<(n + 63)/64, 64, 0, stream>>>(rays, scales, rotsv, opac, shs, timef,
                                             labelW, deformW, ws, out, n);
}
